// BertUnpadSelfAttention_19774029431771
// MI455X (gfx1250) — matrix operand order repaired and checked by reading, not compiled, not run on hardware
//
#include <hip/hip_runtime.h>

// ---------------------------------------------------------------------------
// BERT unpadded self-attention for MI455X (gfx1250, wave32, WMMA).
// Pass 1: zero padded Q/K/V scratch (bf16)       -> matches reference zeros
// Pass 2: bf16 WMMA GEMM qkv = hidden @ W^T + b, scatter via indices
//         (double-buffered LDS, single barrier per K-step)
// Pass 3: flash attention, v_wmma_f32_16x16x32_bf16, fp32 alibi/softmax,
//         double-buffered K/V tiles, async global->LDS K copies (ASYNCcnt).
// ---------------------------------------------------------------------------

#define NBATCH 8
#define SMAX   1024
#define HID    768
#define NHEADS 12
#define HDIM   64

typedef __attribute__((ext_vector_type(16))) __bf16 bf16x16;
typedef __attribute__((ext_vector_type(2)))  __bf16 bf16x2;
typedef __attribute__((ext_vector_type(8)))  float  v8f;
typedef __attribute__((ext_vector_type(4)))  int    v4i;

#define AS1 __attribute__((address_space(1)))
#define AS3 __attribute__((address_space(3)))

#if defined(__has_builtin)
#  if __has_builtin(__builtin_amdgcn_global_load_async_to_lds_b128) && \
      __has_builtin(__builtin_amdgcn_s_wait_asynccnt)
#    define USE_ASYNC_LDS 1
#  endif
#endif
#ifndef USE_ASYNC_LDS
#  define USE_ASYNC_LDS 0
#endif

struct __align__(16) U16B { unsigned x, y, z, w; };
struct __align__(16) F4   { float a, b, c, d; };

__device__ __forceinline__ v8f v8zero() {
  v8f z = {0.f, 0.f, 0.f, 0.f, 0.f, 0.f, 0.f, 0.f};
  return z;
}
__device__ __forceinline__ U16B zero16() {
  U16B z; z.x = 0u; z.y = 0u; z.z = 0u; z.w = 0u;
  return z;
}
__device__ __forceinline__ U16B ldg16(const unsigned short* p) {
  return *(const U16B*)p;
}

// fp32 -> bf16 via hardware convert (RNE)
__device__ __forceinline__ unsigned short f2bf(float x) {
  __bf16 h = (__bf16)x;
  return __builtin_bit_cast(unsigned short, h);
}
// pack two fp32 -> bf16x2 (backend can select v_cvt_pk_bf16_f32)
__device__ __forceinline__ unsigned pk2bf(float x, float y) {
  bf16x2 t;
  t.x = (__bf16)x;
  t.y = (__bf16)y;
  return __builtin_bit_cast(unsigned, t);
}

// A-frag (16-bit): per lane 8 halfs at p, 8 halfs at p+16 (K split pattern)
__device__ __forceinline__ bf16x16 frag_gap8(const unsigned short* p) {
  union { bf16x16 v; U16B q[2]; } f;
  f.q[0] = *(const U16B*)(p);
  f.q[1] = *(const U16B*)(p + 16);
  return f.v;
}
// B-frag (16-bit): per lane 16 contiguous halfs at p
__device__ __forceinline__ bf16x16 frag_c16(const unsigned short* p) {
  union { bf16x16 v; U16B q[2]; } f;
  f.q[0] = *(const U16B*)(p);
  f.q[1] = *(const U16B*)(p + 8);
  return f.v;
}

// async 16B global -> LDS copy (gfx1250 ASYNCcnt path).
// Builtin signature (from hipcc diagnostic): (v4i AS1*, v4i AS3*, imm, imm).
__device__ __forceinline__ void async_copy16(const unsigned short* g,
                                             unsigned short* l) {
#if USE_ASYNC_LDS
  __builtin_amdgcn_global_load_async_to_lds_b128(
      (AS1 v4i*)(unsigned long long)g,
      (AS3 v4i*)(unsigned)(unsigned long long)l, 0, 0);
#else
  (void)g; (void)l;
#endif
}

// convert+stage 16 contiguous fp32 -> 8 packed bf16x2 words
__device__ __forceinline__ void stage16(const float* src, unsigned pk[8]) {
  const F4* f = (const F4*)src;
  F4 q0 = f[0], q1 = f[1], q2 = f[2], q3 = f[3];
  pk[0] = pk2bf(q0.a, q0.b); pk[1] = pk2bf(q0.c, q0.d);
  pk[2] = pk2bf(q1.a, q1.b); pk[3] = pk2bf(q1.c, q1.d);
  pk[4] = pk2bf(q2.a, q2.b); pk[5] = pk2bf(q2.c, q2.d);
  pk[6] = pk2bf(q3.a, q3.b); pk[7] = pk2bf(q3.c, q3.d);
}
__device__ __forceinline__ void store_pk(unsigned short* dst, const unsigned pk[8]) {
  U16B a; a.x = pk[0]; a.y = pk[1]; a.z = pk[2]; a.w = pk[3];
  U16B b; b.x = pk[4]; b.y = pk[5]; b.z = pk[6]; b.w = pk[7];
  ((U16B*)dst)[0] = a;
  ((U16B*)dst)[1] = b;
}

// ---------------------------------------------------------------------------
__global__ __launch_bounds__(256)
void zero_ws_kernel(U16B* __restrict__ p, int n16) {
  int i = blockIdx.x * 256 + threadIdx.x;
  if (i < n16) p[i] = zero16();
}

// ---------------------------------------------------------------------------
// qkv[m,n] = sum_k hidden[m,k]*W[n,k] + b[n], scattered to padded Q/K/V bf16.
// Block tile 128x128, 8 waves of 32(M)x64(N); double-buffered LDS; one
// barrier per K-step; next step's globals staged in registers during WMMA.
// ---------------------------------------------------------------------------
__global__ __launch_bounds__(256)
void qkv_gemm_scatter(const float* __restrict__ hid,
                      const float* __restrict__ W,
                      const float* __restrict__ bias,
                      const long long* __restrict__ idx,
                      int total,
                      unsigned short* __restrict__ Qp,
                      unsigned short* __restrict__ Kp,
                      unsigned short* __restrict__ Vp) {
  __shared__ __align__(16) unsigned short lA[2][128 * 40];
  __shared__ __align__(16) unsigned short lB[2][128 * 40];

  const int ntile = blockIdx.x;          // 0..17
  const int mtile = blockIdx.y;
  const int tid  = threadIdx.x;
  const int w    = tid >> 5;
  const int lane = tid & 31;
  const int wm = w & 3;                  // 4 wave rows (32 M each)
  const int wn = w >> 2;                 // 2 wave cols (64 N each)
  const int hs = lane >> 4;
  const int ln = lane & 15;

  v8f acc[2][4];
#pragma unroll
  for (int i = 0; i < 2; ++i)
#pragma unroll
    for (int j = 0; j < 4; ++j) acc[i][j] = v8zero();

  const int  lrow   = tid >> 1;          // 0..127
  const int  seg    = tid & 1;           // 16-half segment within k-step
  const int  mrow_g = mtile * 128 + lrow;
  const int  nrow_g = ntile * 128 + lrow;
  const bool mvalid = (mrow_g < total);
  const float* aSrc = hid + (long long)mrow_g * HID + seg * 16;
  const float* bSrc = W   + (long long)nrow_g * HID + seg * 16;

  unsigned apk[8], bpk[8];
  // prologue: stage k0 = 0
  if (mvalid) stage16(aSrc, apk);
  else {
#pragma unroll
    for (int p = 0; p < 8; ++p) apk[p] = 0u;
  }
  stage16(bSrc, bpk);

  const int NIT = HID / 32;              // 24
  for (int it = 0; it < NIT; ++it) {
    const int buf = it & 1;
    store_pk(&lA[buf][lrow * 40 + seg * 16], apk);
    store_pk(&lB[buf][lrow * 40 + seg * 16], bpk);
    __syncthreads();

    if (it + 1 < NIT) {                  // stage next step during compute
      const int k0 = (it + 1) * 32;
      if (mvalid) stage16(aSrc + k0, apk);
      else {
#pragma unroll
        for (int p = 0; p < 8; ++p) apk[p] = 0u;
      }
      stage16(bSrc + k0, bpk);
    }

    bf16x16 af[2], bfr[4];
#pragma unroll
    for (int i = 0; i < 2; ++i)
      af[i] = frag_c16(&lA[buf][(wm * 32 + i * 16 + ln) * 40 + hs * 16]);
#pragma unroll
    for (int j = 0; j < 4; ++j)
      bfr[j] = frag_c16(&lB[buf][(wn * 64 + j * 16 + ln) * 40 + hs * 16]);
#pragma unroll
    for (int i = 0; i < 2; ++i)
#pragma unroll
      for (int j = 0; j < 4; ++j)
        acc[i][j] = __builtin_amdgcn_wmma_f32_16x16x32_bf16(
            false, af[i], false, bfr[j], (short)0, acc[i][j], false, false);
  }

  // ---- bias + scatter through indices into padded Q/K/V ----
#pragma unroll
  for (int i = 0; i < 2; ++i) {
#pragma unroll
    for (int r = 0; r < 8; ++r) {
      const int m = mtile * 128 + wm * 32 + i * 16 + r + hs * 8;
      if (m < total) {
        const long long pos = idx[m];
        const int bb = (int)(pos >> 10);
        const int ss = (int)(pos & (SMAX - 1));
#pragma unroll
        for (int j = 0; j < 4; ++j) {
          const int n   = ntile * 128 + wn * 64 + j * 16 + ln;
          const float v = acc[i][j][r] + bias[n];
          const int part = n / HID;            // 0=q 1=k 2=v
          const int rem  = n - part * HID;
          const int hh   = rem >> 6;
          const int dc   = rem & 63;
          unsigned short* dst = (part == 0) ? Qp : ((part == 1) ? Kp : Vp);
          dst[(((long long)bb * NHEADS + hh) * SMAX + ss) * HDIM + dc] = f2bf(v);
        }
      }
    }
  }
}

// ---------------------------------------------------------------------------
// Flash attention: one block per (b, h, 64-row s-tile); 4 waves x 16 rows.
// Double-buffered K [t][d] / V-transposed [d][t] tiles; K via async
// global->LDS (issued one chunk ahead) when available; P per-wave in LDS.
// ---------------------------------------------------------------------------
__global__ __launch_bounds__(128)
void attn_kernel(const unsigned short* __restrict__ Qp,
                 const unsigned short* __restrict__ Kp,
                 const unsigned short* __restrict__ Vp,
                 const float* __restrict__ alibi,
                 const int* __restrict__ cu,
                 float* __restrict__ out) {
  __shared__ __align__(16) unsigned short Kt[2][64 * 72];     // [t][d]
  __shared__ __align__(16) unsigned short Vt[2][64 * 72];     // [d][t]
  __shared__ __align__(16) unsigned short Pt[4][16 * 72];     // per-wave P

  const int stile = blockIdx.x;
  const int h     = blockIdx.y;
  const int b     = blockIdx.z;
  const int cu0   = cu[b];
  const int L     = cu[b + 1] - cu0;
  const int sblk  = stile * 64;
  if (sblk >= L) return;                 // fully padded tile: uniform exit

  const int tid  = threadIdx.x;
  const int w    = tid >> 5;
  const int lane = tid & 31;
  const int hs   = lane >> 4;
  const int ln   = lane & 15;
  const int sbase = sblk + w * 16;
  const long long bh = (long long)b * NHEADS + h;
  const unsigned short* Kbase = Kp + bh * SMAX * HDIM;
  const unsigned short* Vbase = Vp + bh * SMAX * HDIM;
  const float* albase = alibi + (bh * SMAX + sbase) * (long long)SMAX;

  // Q fragments for this wave's 16 rows, registers for the whole loop
  const unsigned short* qrow = Qp + (bh * SMAX + (sbase + ln)) * HDIM;
  bf16x16 aq[2];
#pragma unroll
  for (int c = 0; c < 2; ++c) aq[c] = frag_c16(qrow + c * 32 + hs * 16);

  float mrow[8], lrow[8];
#pragma unroll
  for (int r = 0; r < 8; ++r) { mrow[r] = -3.0e38f; lrow[r] = 0.f; }
  v8f o[4];
#pragma unroll
  for (int d = 0; d < 4; ++d) o[d] = v8zero();

  const int nch = (L + 63) >> 6;

  // V staging registers: 2 units/thread, each = two t-rows x 8 d-values
  U16B v0r[2], v1r[2];
#if !USE_ASYNC_LDS
  U16B kr[4];                            // K staging: 4 x 16B per thread
#endif

  // ---- prologue: stage chunk 0 ----
#pragma unroll
  for (int u = 0; u < 2; ++u) {
    const int uid = (tid << 1) + u, tp = uid >> 3, sg = uid & 7;
    const int t = 2 * tp;
    v0r[u] = (t     < L) ? ldg16(Vbase + (long long)t * HDIM + sg * 8)       : zero16();
    v1r[u] = (t + 1 < L) ? ldg16(Vbase + (long long)(t + 1) * HDIM + sg * 8) : zero16();
  }
#if USE_ASYNC_LDS
#pragma unroll
  for (int it = 0; it < 4; ++it) {
    const int cid = (tid << 2) + it, tl = cid >> 3, part = cid & 7;
    unsigned short* d = &Kt[0][tl * 72 + part * 8];
    if (tl < L) async_copy16(Kbase + (long long)tl * HDIM + part * 8, d);
    else        *(U16B*)d = zero16();
  }
#else
#pragma unroll
  for (int it = 0; it < 4; ++it) {
    const int cid = (tid << 2) + it, tl = cid >> 3, part = cid & 7;
    kr[it] = (tl < L) ? ldg16(Kbase + (long long)tl * HDIM + part * 8) : zero16();
  }
#endif

  for (int ch = 0; ch < nch; ++ch) {
    const int t0  = ch * 64;
    const int buf = ch & 1;

    // ---- commit staged V (transposed) and K into buf ----
#pragma unroll
    for (int u = 0; u < 2; ++u) {
      const int uid = (tid << 1) + u, tp = uid >> 3, sg = uid & 7;
      const unsigned a[4] = {v0r[u].x, v0r[u].y, v0r[u].z, v0r[u].w};
      const unsigned c[4] = {v1r[u].x, v1r[u].y, v1r[u].z, v1r[u].w};
#pragma unroll
      for (int e = 0; e < 8; ++e) {
        const unsigned lo = (a[e >> 1] >> ((e & 1) * 16)) & 0xffffu;
        const unsigned hi = (c[e >> 1] >> ((e & 1) * 16)) << 16;
        *(unsigned*)&Vt[buf][(sg * 8 + e) * 72 + 2 * tp] = lo | hi;
      }
    }
#if USE_ASYNC_LDS
    __builtin_amdgcn_s_wait_asynccnt(0);
#else
#pragma unroll
    for (int it = 0; it < 4; ++it) {
      const int cid = (tid << 2) + it, tl = cid >> 3, part = cid & 7;
      *(U16B*)&Kt[buf][tl * 72 + part * 8] = kr[it];
    }
#endif
    __syncthreads();

    // ---- stage next chunk while this chunk computes ----
    if (ch + 1 < nch) {
      const int t0n = (ch + 1) * 64;
#if USE_ASYNC_LDS
#pragma unroll
      for (int it = 0; it < 4; ++it) {
        const int cid = (tid << 2) + it, tl = cid >> 3, part = cid & 7;
        const int tg  = t0n + tl;
        unsigned short* d = &Kt[buf ^ 1][tl * 72 + part * 8];
        if (tg < L) async_copy16(Kbase + (long long)tg * HDIM + part * 8, d);
        else        *(U16B*)d = zero16();
      }
#else
#pragma unroll
      for (int it = 0; it < 4; ++it) {
        const int cid = (tid << 2) + it, tl = cid >> 3, part = cid & 7;
        const int tg  = t0n + tl;
        kr[it] = (tg < L) ? ldg16(Kbase + (long long)tg * HDIM + part * 8) : zero16();
      }
#endif
#pragma unroll
      for (int u = 0; u < 2; ++u) {
        const int uid = (tid << 1) + u, tp = uid >> 3, sg = uid & 7;
        const int t = t0n + 2 * tp;
        v0r[u] = (t     < L) ? ldg16(Vbase + (long long)t * HDIM + sg * 8)       : zero16();
        v1r[u] = (t + 1 < L) ? ldg16(Vbase + (long long)(t + 1) * HDIM + sg * 8) : zero16();
      }
      // prefetch next alibi tile (16 rows x 64 cols fp32, one line per lane)
      __builtin_prefetch(albase + (long long)(lane >> 1) * SMAX + t0n + (lane & 1) * 32, 0, 0);
    }

    // ---- scores S = Q K^T * scale + alibi (fp32 accum) ----
    v8f sc[4];
#pragma unroll
    for (int ts = 0; ts < 4; ++ts) {
      v8f s = v8zero();
#pragma unroll
      for (int c = 0; c < 2; ++c) {
        bf16x16 bk = frag_c16(&Kt[buf][(ts * 16 + ln) * 72 + c * 32 + hs * 16]);
        s = __builtin_amdgcn_wmma_f32_16x16x32_bf16(
            false, aq[c], false, bk, (short)0, s, false, false);
      }
      const float* al = albase + (t0 + ts * 16 + ln);
#pragma unroll
      for (int r = 0; r < 8; ++r) {
        const int rl = r + hs * 8;
        s[r] = s[r] * 0.125f + al[(long long)rl * SMAX];
      }
      sc[ts] = s;
    }

    // ---- online softmax (16-lane half reductions, wave32) ----
#pragma unroll
    for (int r = 0; r < 8; ++r) {
      float cm = fmaxf(fmaxf(sc[0][r], sc[1][r]), fmaxf(sc[2][r], sc[3][r]));
#pragma unroll
      for (int m = 1; m < 16; m <<= 1) cm = fmaxf(cm, __shfl_xor(cm, m, 32));
      const float nm   = fmaxf(mrow[r], cm);
      const float corr = __expf(mrow[r] - nm);
      mrow[r] = nm;
      const int rl = r + hs * 8;
      float ps = 0.f;
#pragma unroll
      for (int ts = 0; ts < 4; ++ts) {
        const float p = __expf(sc[ts][r] - nm);
        ps += p;
        Pt[w][rl * 72 + ts * 16 + ln] = f2bf(p);
      }
#pragma unroll
      for (int m = 1; m < 16; m <<= 1) ps += __shfl_xor(ps, m, 32);
      lrow[r] = lrow[r] * corr + ps;
#pragma unroll
      for (int d = 0; d < 4; ++d) o[d][r] *= corr;
    }

    // ---- O += P V (P via per-wave LDS round-trip into A-frag layout) ----
#pragma unroll
    for (int c = 0; c < 2; ++c) {
      bf16x16 ap = frag_c16(&Pt[w][ln * 72 + c * 32 + hs * 16]);
#pragma unroll
      for (int d = 0; d < 4; ++d) {
        bf16x16 bv = frag_c16(&Vt[buf][(d * 16 + ln) * 72 + c * 32 + hs * 16]);
        o[d] = __builtin_amdgcn_wmma_f32_16x16x32_bf16(
            false, ap, false, bv, (short)0, o[d], false, false);
      }
    }
  }

  // ---- normalize and gather-store valid rows ----
#pragma unroll
  for (int r = 0; r < 8; ++r) {
    const int srow = sbase + r + hs * 8;
    if (srow < L) {
      const float inv = 1.0f / lrow[r];
      float* orow = out + (long long)(cu0 + srow) * HID + h * HDIM;
#pragma unroll
      for (int d = 0; d < 4; ++d) orow[d * 16 + ln] = o[d][r] * inv;
    }
  }
}

// ---------------------------------------------------------------------------
extern "C" void kernel_launch(void* const* d_in, const int* in_sizes, int n_in,
                              void* d_out, int out_size, void* d_ws, size_t ws_size,
                              hipStream_t stream) {
  const float*     hidden = (const float*)d_in[0];
  const float*     W      = (const float*)d_in[1];
  const float*     bias   = (const float*)d_in[2];
  const float*     alibi  = (const float*)d_in[3];
  const int*       cu     = (const int*)d_in[4];
  const long long* idx    = (const long long*)d_in[6];
  const int total = in_sizes[0] / HID;

  const long long qkv_elems = (long long)NBATCH * NHEADS * SMAX * HDIM; // 6291456
  unsigned short* Qp = (unsigned short*)d_ws;
  unsigned short* Kp = Qp + qkv_elems;
  unsigned short* Vp = Kp + qkv_elems;

  // zero padded Q/K/V (bf16) so padded rows are exact zeros like the reference
  const int n16 = (int)(3 * qkv_elems / 8);   // 16-byte vectors
  zero_ws_kernel<<<(n16 + 255) / 256, 256, 0, stream>>>((U16B*)d_ws, n16);

  dim3 gg(3 * HID / 128, (total + 127) / 128);
  qkv_gemm_scatter<<<gg, 256, 0, stream>>>(hidden, W, bias, idx, total, Qp, Kp, Vp);

  dim3 ga(SMAX / 64, NHEADS, NBATCH);
  attn_kernel<<<ga, 128, 0, stream>>>(Qp, Kp, Vp, alibi, cu, (float*)d_out);
}
